// TSMixerC_20779051778728
// MI455X (gfx1250) — matrix operand order repaired and checked by reading, not compiled, not run on hardware
//
#include <hip/hip_runtime.h>
#include <math.h>

// ---------------------------------------------------------------------------
// TSMixer forward for MI455X (gfx1250), bf16 WMMA path, fp32 accumulate.
// ---------------------------------------------------------------------------

typedef __bf16 bf16_t;
typedef __attribute__((ext_vector_type(16))) __bf16 v16bf;
typedef __attribute__((ext_vector_type(8)))  __bf16 v8bf;
typedef __attribute__((ext_vector_type(8)))  float  v8f;

#define DEVFN static __device__ __forceinline__

constexpr int Bn = 32, Sn = 512, Vn = 512, On = 336, En = 8, Ln = 4, Fn = 2048;
constexpr float EPSv = 1e-5f;

union V16U { v16bf v; v8bf h[2]; };

DEVFN v8f wmma_bf16(v16bf a, v16bf b, v8f c) {
  // D = A(16x32) * B(32x16) + C, fp32 accum
  return __builtin_amdgcn_wmma_f32_16x16x32_bf16(false, a, false, b, (short)0, c,
                                                 false, false);
}

// A fragment: lane holds row (lane&15); kbase already includes (lane>>4)*8.
// VGPR0-3 = K kbase..kbase+7, VGPR4-7 = K kbase+16..kbase+23 (per ISA layout).
DEVFN v16bf load_a_frag(const bf16_t* base, int ld, int row, int kbase) {
  const bf16_t* p = base + (size_t)row * ld + kbase;
  V16U r;
  r.h[0] = *(const v8bf*)p;
  r.h[1] = *(const v8bf*)(p + 16);
  return r.v;
}

// B fragment: lane holds column n0+(lane&15); elements 0-7 = K kb..kb+7,
// 8-15 = K kb+16..kb+23, kb = step base + (lane>>4)*8 (same as A).
DEVFN v16bf load_b_frag(const bf16_t* base, int ld, int k, int n0) {
  int l = k & 31;
  int kb = (k - l) + (l >> 4) * 8;
  const bf16_t* p = base + n0 + (l & 15);
  v16bf r;
#pragma unroll
  for (int s = 0; s < 16; ++s)
    r[s] = p[(size_t)(kb + (s < 8 ? s : s + 8)) * ld];
  return r;
}

// ---------------------------------------------------------------------------
// Prep kernels
// ---------------------------------------------------------------------------

__global__ __launch_bounds__(256) void k_cvt(const float* __restrict__ src,
                                             bf16_t* __restrict__ dst) {
  size_t i = (size_t)blockIdx.x * 256 + threadIdx.x;
  dst[i] = (bf16_t)src[i];
}

// src: [nmat][R][C] f32 -> dst: [nmat][C][R] bf16 (flat over nmat*C*R outputs)
__global__ __launch_bounds__(256) void k_trans(const float* __restrict__ src,
                                               bf16_t* __restrict__ dst,
                                               int R, int C) {
  size_t i = (size_t)blockIdx.x * 256 + threadIdx.x;
  size_t per = (size_t)R * C;
  size_t l = i / per;
  size_t rem = i - l * per;
  int c = (int)(rem / R);
  int r = (int)(rem - (size_t)c * R);
  dst[i] = (bf16_t)src[l * per + (size_t)r * C + c];
}

// ---------------------------------------------------------------------------
// RevIN stats + normalize
// ---------------------------------------------------------------------------

__global__ __launch_bounds__(256) void k_stats(const float* __restrict__ x,
                                               float* __restrict__ mean,
                                               float* __restrict__ stdv) {
  int i = blockIdx.x * 256 + threadIdx.x;     // over B*V
  int b = i / Vn, v = i - b * Vn;
  const float* p = x + (size_t)b * Sn * Vn + v;
  float s = 0.f, sq = 0.f;
  for (int t = 0; t < Sn; ++t) {
    float val = p[(size_t)t * Vn];
    s += val;
    sq += val * val;
  }
  float mu = s * (1.f / Sn);
  float var = sq * (1.f / Sn) - mu * mu;
  mean[i] = mu;
  stdv[i] = sqrtf(var + EPSv);
}

__global__ __launch_bounds__(256) void k_revin(const float* __restrict__ x,
                                               const float* __restrict__ rev_w,
                                               const float* __restrict__ rev_b,
                                               const float* __restrict__ mean,
                                               const float* __restrict__ stdv,
                                               float* __restrict__ h,
                                               bf16_t* __restrict__ hbf) {
  size_t i = (size_t)blockIdx.x * 256 + threadIdx.x;   // over B*S*V
  int v = (int)(i % Vn);
  int b = (int)(i / ((size_t)Sn * Vn));
  int bv = b * Vn + v;
  float val = (x[i] - mean[bv]) / stdv[bv] * rev_w[v] + rev_b[v];
  h[i] = val;
  hbf[i] = (bf16_t)val;
}

// ---------------------------------------------------------------------------
// MoE gate: softmax over E=8, top-2 -> dense combine weights wd[B,V,E]
// ---------------------------------------------------------------------------

__global__ __launch_bounds__(256) void k_gate(const float* __restrict__ h,
                                              const float* __restrict__ gw,
                                              float* __restrict__ wd) {
  __shared__ float gls[En * Sn];               // 16 KB
  int tid = threadIdx.x;
  for (int i = tid; i < En * Sn; i += 256) gls[i] = gw[i];
  __syncthreads();
  int b = blockIdx.y;
  int v = blockIdx.x * 256 + tid;
  const float* hp = h + (size_t)b * Sn * Vn + v;
  float a[En];
#pragma unroll
  for (int e = 0; e < En; ++e) a[e] = 0.f;
  for (int s = 0; s < Sn; ++s) {
    float xv = hp[(size_t)s * Vn];
#pragma unroll
    for (int e = 0; e < En; ++e) a[e] = fmaf(xv, gls[e * Sn + s], a[e]);
  }
  float mx = a[0];
#pragma unroll
  for (int e = 1; e < En; ++e) mx = fmaxf(mx, a[e]);
  float sum = 0.f;
#pragma unroll
  for (int e = 0; e < En; ++e) { a[e] = __expf(a[e] - mx); sum += a[e]; }
  float inv = 1.f / sum;
#pragma unroll
  for (int e = 0; e < En; ++e) a[e] *= inv;
  int i0 = 0;
#pragma unroll
  for (int e = 1; e < En; ++e) if (a[e] > a[i0]) i0 = e;
  int i1 = (i0 == 0) ? 1 : 0;
#pragma unroll
  for (int e = 0; e < En; ++e) if (e != i0 && a[e] > a[i1]) i1 = e;
  float* o = wd + ((size_t)b * Vn + v) * En;
#pragma unroll
  for (int e = 0; e < En; ++e)
    o[e] = (e == i0) ? a[i0] : ((e == i1) ? a[i1] : 0.f);
}

// ---------------------------------------------------------------------------
// Time-mixing MoE: D[o][v] tile = sum_e wd * (hbf[b,:,v] x ew[e,o,:]) + eb
// then relu + residual into h (and bf16 shadow hbf_out). One wave per tile.
// ---------------------------------------------------------------------------

__global__ __launch_bounds__(32) void k_timemix(const bf16_t* __restrict__ hbf,
                                                const bf16_t* __restrict__ ew,
                                                const float* __restrict__ eb,
                                                const float* __restrict__ wd,
                                                float* __restrict__ h,
                                                bf16_t* __restrict__ hbf_out) {
  const int lane = threadIdx.x & 31;
  const int v0 = blockIdx.x * 16, o0 = blockIdx.y * 16, b = blockIdx.z;
  const int m = lane & 15, khalf = lane >> 4;
  v8f zero = {};
  v8f acc[En];
#pragma unroll
  for (int e = 0; e < En; ++e) acc[e] = zero;
  const bf16_t* bb = hbf + (size_t)b * Sn * Vn + v0;
#pragma unroll 2
  for (int kk = 0; kk < Sn / 32; ++kk) {
    v16bf bfr = load_b_frag(bb, Vn, kk * 32 + lane, 0);
#pragma unroll
    for (int e = 0; e < En; ++e) {
      v16bf afr = load_a_frag(ew + (size_t)e * Sn * Sn, Sn, o0 + m,
                              kk * 32 + khalf * 8);
      acc[e] = wmma_bf16(afr, bfr, acc[e]);
    }
  }
  const float* wdp = wd + ((size_t)b * Vn + v0 + m) * En;
  float4 wA = *(const float4*)wdp;
  float4 wB = *(const float4*)(wdp + 4);
  float wde[8] = {wA.x, wA.y, wA.z, wA.w, wB.x, wB.y, wB.z, wB.w};
  float comb[8];
#pragma unroll
  for (int r = 0; r < 8; ++r) comb[r] = 0.f;
#pragma unroll
  for (int e = 0; e < En; ++e) {
    const float* ebp = eb + e * Sn + o0 + khalf * 8;
    float4 e0 = *(const float4*)ebp;
    float4 e1 = *(const float4*)(ebp + 4);
    float ebv[8] = {e0.x, e0.y, e0.z, e0.w, e1.x, e1.y, e1.z, e1.w};
#pragma unroll
    for (int r = 0; r < 8; ++r)
      comb[r] = fmaf(wde[e], acc[e][r] + ebv[r], comb[r]);
  }
#pragma unroll
  for (int r = 0; r < 8; ++r) {
    int o = o0 + khalf * 8 + r;
    size_t idx = (size_t)b * Sn * Vn + (size_t)o * Vn + v0 + m;
    float val = fmaxf(comb[r], 0.f) + h[idx];
    h[idx] = val;
    hbf_out[idx] = (bf16_t)val;
  }
}

// ---------------------------------------------------------------------------
// Feature-mixing MLP: fc1 (V->F, relu, bf16 out), fc2 (F->V, +bias+residual)
// Weights pre-transposed to K-major. 4 N-tiles per wave, streaming-A.
// ---------------------------------------------------------------------------

__global__ __launch_bounds__(32) void k_fc1(const bf16_t* __restrict__ A,
                                            const bf16_t* __restrict__ Bm,
                                            const float* __restrict__ bias,
                                            bf16_t* __restrict__ outbf) {
  const int lane = threadIdx.x & 31;
  const int n0 = blockIdx.x * 64, t0 = blockIdx.y * 16;
  const int m = lane & 15, khalf = lane >> 4;
  v8f zero = {};
  v8f acc[4];
#pragma unroll
  for (int j = 0; j < 4; ++j) acc[j] = zero;
#pragma unroll 2
  for (int kk = 0; kk < Vn / 32; ++kk) {
    v16bf afr = load_a_frag(A + (size_t)t0 * Vn, Vn, m, kk * 32 + khalf * 8);
#pragma unroll
    for (int j = 0; j < 4; ++j) {
      v16bf bfr = load_b_frag(Bm, Fn, kk * 32 + lane, n0 + j * 16);
      acc[j] = wmma_bf16(afr, bfr, acc[j]);
    }
  }
#pragma unroll
  for (int j = 0; j < 4; ++j) {
    int f = n0 + j * 16 + m;
    float bi = bias[f];
#pragma unroll
    for (int r = 0; r < 8; ++r) {
      int t = t0 + khalf * 8 + r;
      float val = fmaxf(acc[j][r] + bi, 0.f);
      outbf[(size_t)t * Fn + f] = (bf16_t)val;
    }
  }
}

__global__ __launch_bounds__(32) void k_fc2(const bf16_t* __restrict__ A,
                                            const bf16_t* __restrict__ Bm,
                                            const float* __restrict__ bias,
                                            float* __restrict__ h,
                                            bf16_t* __restrict__ hbf) {
  const int lane = threadIdx.x & 31;
  const int n0 = blockIdx.x * 64, t0 = blockIdx.y * 16;
  const int m = lane & 15, khalf = lane >> 4;
  v8f zero = {};
  v8f acc[4];
#pragma unroll
  for (int j = 0; j < 4; ++j) acc[j] = zero;
#pragma unroll 2
  for (int kk = 0; kk < Fn / 32; ++kk) {
    v16bf afr = load_a_frag(A + (size_t)t0 * Fn, Fn, m, kk * 32 + khalf * 8);
#pragma unroll
    for (int j = 0; j < 4; ++j) {
      v16bf bfr = load_b_frag(Bm, Vn, kk * 32 + lane, n0 + j * 16);
      acc[j] = wmma_bf16(afr, bfr, acc[j]);
    }
  }
#pragma unroll
  for (int j = 0; j < 4; ++j) {
    int v = n0 + j * 16 + m;
    float bi = bias[v];
#pragma unroll
    for (int r = 0; r < 8; ++r) {
      int t = t0 + khalf * 8 + r;
      size_t idx = (size_t)t * Vn + v;
      float val = acc[j][r] + bi + h[idx];
      h[idx] = val;
      hbf[idx] = (bf16_t)val;
    }
  }
}

// ---------------------------------------------------------------------------
// Projection MoE (S->O=336) + RevIN denorm, writes final output [B,O,V]
// ---------------------------------------------------------------------------

__global__ __launch_bounds__(32) void k_proj(const bf16_t* __restrict__ hbf,
                                             const bf16_t* __restrict__ pw,
                                             const float* __restrict__ pb,
                                             const float* __restrict__ wd,
                                             const float* __restrict__ rev_w,
                                             const float* __restrict__ rev_b,
                                             const float* __restrict__ mean,
                                             const float* __restrict__ stdv,
                                             float* __restrict__ out) {
  const int lane = threadIdx.x & 31;
  const int v0 = blockIdx.x * 16, o0 = blockIdx.y * 16, b = blockIdx.z;
  const int m = lane & 15, khalf = lane >> 4;
  v8f zero = {};
  v8f acc[En];
#pragma unroll
  for (int e = 0; e < En; ++e) acc[e] = zero;
  const bf16_t* bb = hbf + (size_t)b * Sn * Vn + v0;
#pragma unroll 2
  for (int kk = 0; kk < Sn / 32; ++kk) {
    v16bf bfr = load_b_frag(bb, Vn, kk * 32 + lane, 0);
#pragma unroll
    for (int e = 0; e < En; ++e) {
      v16bf afr = load_a_frag(pw + (size_t)e * On * Sn, Sn, o0 + m,
                              kk * 32 + khalf * 8);
      acc[e] = wmma_bf16(afr, bfr, acc[e]);
    }
  }
  const float* wdp = wd + ((size_t)b * Vn + v0 + m) * En;
  float4 wA = *(const float4*)wdp;
  float4 wB = *(const float4*)(wdp + 4);
  float wde[8] = {wA.x, wA.y, wA.z, wA.w, wB.x, wB.y, wB.z, wB.w};
  float comb[8];
#pragma unroll
  for (int r = 0; r < 8; ++r) comb[r] = 0.f;
#pragma unroll
  for (int e = 0; e < En; ++e) {
    const float* ebp = pb + e * On + o0 + khalf * 8;
    float4 e0 = *(const float4*)ebp;
    float4 e1 = *(const float4*)(ebp + 4);
    float ebv[8] = {e0.x, e0.y, e0.z, e0.w, e1.x, e1.y, e1.z, e1.w};
#pragma unroll
    for (int r = 0; r < 8; ++r)
      comb[r] = fmaf(wde[e], acc[e][r] + ebv[r], comb[r]);
  }
  int v = v0 + m;
  float rb = rev_b[v];
  float rwi = 1.f / (rev_w[v] + EPSv * EPSv);
  float sd = stdv[b * Vn + v];
  float mu = mean[b * Vn + v];
#pragma unroll
  for (int r = 0; r < 8; ++r) {
    int o = o0 + khalf * 8 + r;
    out[(size_t)b * On * Vn + (size_t)o * Vn + v] =
        (comb[r] - rb) * rwi * sd + mu;
  }
}

// ---------------------------------------------------------------------------
// Host launcher
// ---------------------------------------------------------------------------

extern "C" void kernel_launch(void* const* d_in, const int* in_sizes, int n_in,
                              void* d_out, int out_size, void* d_ws,
                              size_t ws_size, hipStream_t stream) {
  (void)in_sizes; (void)n_in; (void)out_size; (void)ws_size;
  const float* x      = (const float*)d_in[0];
  const float* rev_w  = (const float*)d_in[1];
  const float* rev_b  = (const float*)d_in[2];
  const float* gate_t = (const float*)d_in[3];
  const float* exp_w  = (const float*)d_in[4];
  const float* exp_b  = (const float*)d_in[5];
  const float* fc1_w  = (const float*)d_in[6];
  const float* fc1_b  = (const float*)d_in[7];
  const float* fc2_w  = (const float*)d_in[8];
  const float* fc2_b  = (const float*)d_in[9];
  const float* gate_p = (const float*)d_in[10];
  const float* proj_w = (const float*)d_in[11];
  const float* proj_b = (const float*)d_in[12];
  float* out = (float*)d_out;

  char* w = (char*)d_ws;
  auto carve = [&](size_t bytes) -> void* {
    void* p = (void*)w;
    w += (bytes + 255) & ~(size_t)255;
    return p;
  };
  float*  mean = (float*)carve((size_t)Bn * Vn * 4);
  float*  stdv = (float*)carve((size_t)Bn * Vn * 4);
  float*  h    = (float*)carve((size_t)Bn * Sn * Vn * 4);
  bf16_t* hbf  = (bf16_t*)carve((size_t)Bn * Sn * Vn * 2);
  bf16_t* hbf2 = (bf16_t*)carve((size_t)Bn * Sn * Vn * 2);
  float*  wd   = (float*)carve((size_t)Bn * Vn * En * 4);
  bf16_t* expw = (bf16_t*)carve((size_t)Ln * En * Sn * Sn * 2);
  bf16_t* pwb  = (bf16_t*)carve((size_t)En * On * Sn * 2);
  bf16_t* fc1T = (bf16_t*)carve((size_t)Ln * Vn * Fn * 2);
  bf16_t* fc2T = (bf16_t*)carve((size_t)Ln * Fn * Vn * 2);
  bf16_t* act1 = (bf16_t*)carve((size_t)Bn * Sn * Fn * 2);

  // one-time (per launch) weight conversion / transposition to bf16
  k_cvt<<<(Ln * En * Sn * Sn) / 256, 256, 0, stream>>>(exp_w, expw);
  k_cvt<<<(En * On * Sn) / 256, 256, 0, stream>>>(proj_w, pwb);
  k_trans<<<(Ln * Fn * Vn) / 256, 256, 0, stream>>>(fc1_w, fc1T, Fn, Vn);
  k_trans<<<(Ln * Vn * Fn) / 256, 256, 0, stream>>>(fc2_w, fc2T, Vn, Fn);

  // RevIN
  k_stats<<<(Bn * Vn) / 256, 256, 0, stream>>>(x, mean, stdv);
  k_revin<<<(Bn * Sn * Vn) / 256, 256, 0, stream>>>(x, rev_w, rev_b, mean, stdv,
                                                    h, hbf);
  for (int l = 0; l < Ln; ++l) {
    k_gate<<<dim3(Vn / 256, Bn), 256, 0, stream>>>(
        h, gate_t + (size_t)l * En * Sn, wd);
    k_timemix<<<dim3(Vn / 16, Sn / 16, Bn), 32, 0, stream>>>(
        hbf, expw + (size_t)l * En * Sn * Sn, exp_b + (size_t)l * En * Sn, wd,
        h, hbf2);
    k_fc1<<<dim3(Fn / 64, (Bn * Sn) / 16), 32, 0, stream>>>(
        hbf2, fc1T + (size_t)l * Vn * Fn, fc1_b + (size_t)l * Fn, act1);
    k_fc2<<<dim3(Vn / 64, (Bn * Sn) / 16), 32, 0, stream>>>(
        act1, fc2T + (size_t)l * Fn * Vn, fc2_b + (size_t)l * Vn, h, hbf);
  }
  k_gate<<<dim3(Vn / 256, Bn), 256, 0, stream>>>(h, gate_p, wd);
  k_proj<<<dim3(Vn / 16, On / 16, Bn), 32, 0, stream>>>(
      hbf, pwb, proj_b, wd, rev_w, rev_b, mean, stdv, out);
}
